// LatentFoldMemory_47467978555546
// MI455X (gfx1250) — compile-verified, matrix operand order checked by reading, not run on hardware
//
#include <hip/hip_runtime.h>
#include <hip/hip_bf16.h>
#include <math.h>

typedef __bf16 bf16_t;
typedef __attribute__((ext_vector_type(16))) __bf16 v16bf;
typedef __attribute__((ext_vector_type(8)))  float  v8f;

#define TOKENS 8192      // B*S
#define DMODEL 2048
#define SEQ    2048
#define BATCH  4
#define NCOL   384       // padded combined projection columns (380 used)

// ---------- helpers ----------

__device__ inline bf16_t f2bf(float f) {
    union { float f; unsigned u; } x; x.f = f;
    unsigned r = x.u + 0x7FFFu + ((x.u >> 16) & 1u);   // round-to-nearest-even
    unsigned short h = (unsigned short)(r >> 16);
    return __builtin_bit_cast(bf16_t, h);
}

__device__ inline float sigmoidf_(float x) { return 1.0f / (1.0f + __expf(-x)); }

// A-fragment (16x32 bf16, M x K), row-major source, stride lda (elements).
// ISA layout: lanes 0-15 -> M=lane, lanes 16-31 -> M=lane-16;
// VGPR j holds K pair base = (j&3)*2 + half*8 + (j>>2)*16.
__device__ inline v16bf load_frag16x32(const bf16_t* p, int lda, int lane) {
    int half = (lane >> 4) & 1;
    int m    = lane & 15;
    v16bf a;
#pragma unroll
    for (int j = 0; j < 8; ++j) {
        int kb = ((j & 3) * 2) + (half * 8) + ((j >> 2) * 16);
        a[2*j]   = p[(size_t)m * lda + kb];
        a[2*j+1] = p[(size_t)m * lda + kb + 1];
    }
    return a;
}

// C/D store: VGPR r, lanes 0-15: M=r,   N=lane; lanes 16-31: M=r+8, N=lane-16.
__device__ inline void store_tileC(float* C, int ldc, int lane, v8f c) {
    int half = (lane >> 4) & 1;
    int n    = lane & 15;
#pragma unroll
    for (int r = 0; r < 8; ++r)
        C[(size_t)(r + 8*half) * ldc + n] = c[r];
}

// ---------- prep kernels ----------

__global__ void convert_x_kernel(const float* __restrict__ x, bf16_t* __restrict__ xb, int n) {
    int i = blockIdx.x * blockDim.x + threadIdx.x;
    if (i < n) xb[i] = f2bf(x[i]);
}

__global__ void pack_wall_kernel(const float* Wk, const float* Wv, const float* Wq,
                                 const float* g0, const float* g1, const float* g2, const float* g3,
                                 const float* pw1, const float* cw,
                                 bf16_t* __restrict__ Wall) {
    int idx = blockIdx.x * blockDim.x + threadIdx.x;
    if (idx >= NCOL * DMODEL) return;
    int n = idx / DMODEL, k = idx % DMODEL;
    float val = 0.0f;
    if      (n < 64)  val = Wk [(n      ) * DMODEL + k];
    else if (n < 128) val = Wv [(n -  64) * DMODEL + k];
    else if (n < 192) val = Wq [(n - 128) * DMODEL + k];
    else if (n < 256) val = g0 [(n - 192) * DMODEL + k];
    else if (n < 288) val = g1 [(n - 256) * DMODEL + k];
    else if (n < 304) val = g2 [(n - 288) * DMODEL + k];
    else if (n < 312) val = g3 [(n - 304) * DMODEL + k];
    else if (n < 376) val = pw1[(n - 312) * DMODEL + k];
    else if (n < 380) val = cw [(n - 376) * DMODEL + k];
    Wall[idx] = f2bf(val);
}

__global__ void pack_wo_kernel(const float* __restrict__ Wo, bf16_t* __restrict__ Wob, int n) {
    int i = blockIdx.x * blockDim.x + threadIdx.x;
    if (i < n) Wob[i] = f2bf(Wo[i]);
}

// ---------- WMMA GEMM:  C[M,N] = A[M,K] * B[N,K]^T  (bf16 in, f32 out) ----------

__global__ __launch_bounds__(256)
void gemm_bf16_kernel(const bf16_t* __restrict__ A, const bf16_t* __restrict__ B,
                      float* __restrict__ C, int M, int N, int K) {
    int wave = (int)((blockIdx.x * blockDim.x + threadIdx.x) >> 5);
    int lane = threadIdx.x & 31;
    int ntn  = N >> 4;
    int tm   = wave / ntn;
    int tn   = wave % ntn;
    if (tm * 16 >= M) return;
    const bf16_t* Ap = A + (size_t)tm * 16 * K;
    const bf16_t* Bp = B + (size_t)tn * 16 * K;
    v8f acc = {};
    for (int k0 = 0; k0 < K; k0 += 32) {
        if (k0 + 32 < K) {  // gfx1250 global_prefetch for next K tile
            __builtin_prefetch(Ap + (size_t)(lane & 15) * K + k0 + 32, 0, 1);
            __builtin_prefetch(Bp + (size_t)(lane & 15) * K + k0 + 32, 0, 1);
        }
        v16bf a = load_frag16x32(Ap + k0, K, lane);
        v16bf b = load_frag16x32(Bp + k0, K, lane);
        acc = __builtin_amdgcn_wmma_f32_16x16x32_bf16(false, a, false, b,
                                                      (short)0, acc, false, false);
    }
    store_tileC(C + (size_t)tm * 16 * N + tn * 16, N, lane, acc);
}

// ---------- per-token elementwise postprocess ----------
// P[t, 0:64]=k_raw  [64:128]=v  [128:192]=q_raw  [192:312]=fold gate pre-act
// [312:376]=promo h pre-act  [376:380]=comb logits

__global__ __launch_bounds__(128)
void postproc_kernel(const float* __restrict__ P,
                     const float* fgb0, const float* fgb1, const float* fgb2, const float* fgb3,
                     const float* lam0, const float* lam1, const float* lam2, const float* lam3,
                     const float* promo_b1, const float* promo_w2, const float* promo_b2,
                     const float* comb_b,
                     float* __restrict__ kn, float* __restrict__ vv, float* __restrict__ qn,
                     float* __restrict__ alphas, float* __restrict__ promo, float* __restrict__ fw) {
    __shared__ float red[128];
    __shared__ float hbuf[64];
    int t = blockIdx.x, tid = threadIdx.x;
    const float* Pt = P + (size_t)t * NCOL;

    // k normalize
    float kvx = (tid < 64) ? Pt[tid] : 0.0f;
    red[tid] = kvx * kvx; __syncthreads();
    for (int o = 64; o > 0; o >>= 1) { if (tid < o) red[tid] += red[tid + o]; __syncthreads(); }
    float knorm = fmaxf(sqrtf(red[0]), 1e-12f); __syncthreads();
    if (tid < 64) kn[(size_t)t * 64 + tid] = kvx / knorm;

    // v copy
    if (tid < 64) vv[(size_t)t * 64 + tid] = Pt[64 + tid];

    // q normalize
    float qvx = (tid < 64) ? Pt[128 + tid] : 0.0f;
    red[tid] = qvx * qvx; __syncthreads();
    for (int o = 64; o > 0; o >>= 1) { if (tid < o) red[tid] += red[tid + o]; __syncthreads(); }
    float qnorm = fmaxf(sqrtf(red[0]), 1e-12f); __syncthreads();
    if (tid < 64) qn[(size_t)t * 64 + tid] = qvx / qnorm;

    // alphas: alpha = exp(8 * sigmoid(gate) * log(sigmoid(lambda)+1e-8))
    if (tid < 120) {
        int f, off;
        if      (tid < 64)  { f = 0; off = tid;       }
        else if (tid < 96)  { f = 1; off = tid - 64;  }
        else if (tid < 112) { f = 2; off = tid - 96;  }
        else                { f = 3; off = tid - 112; }
        const float* fb  = (f == 0) ? fgb0 : (f == 1) ? fgb1 : (f == 2) ? fgb2 : fgb3;
        const float* lam = (f == 0) ? lam0 : (f == 1) ? lam1 : (f == 2) ? lam2 : lam3;
        float r  = sigmoidf_(Pt[192 + tid] + fb[off]);
        float la = logf(sigmoidf_(lam[off]) + 1e-8f);
        alphas[(size_t)t * 120 + tid] = __expf(8.0f * r * la);
    }

    // promotion gate: gelu(exact) then 3x64 matvec + sigmoid
    if (tid < 64) {
        float h = Pt[312 + tid] + promo_b1[tid];
        hbuf[tid] = 0.5f * h * (1.0f + erff(h * 0.70710678118654752f));
    }
    __syncthreads();
    if (tid < 3) {
        float s = promo_b2[tid];
        for (int j = 0; j < 64; ++j) s += hbuf[j] * promo_w2[tid * 64 + j];
        promo[(size_t)t * 3 + tid] = sigmoidf_(s);
    }

    // fold-combiner softmax (4-way)
    if (tid == 0) {
        float l[4], mx = -1e30f, se = 0.0f;
        for (int i = 0; i < 4; ++i) { l[i] = Pt[376 + i] + comb_b[i]; mx = fmaxf(mx, l[i]); }
        for (int i = 0; i < 4; ++i) { l[i] = __expf(l[i] - mx); se += l[i]; }
        for (int i = 0; i < 4; ++i) fw[(size_t)t * 4 + i] = l[i] / se;
    }
}

// ---------- the sequential scan: one workgroup per batch, H in LDS ----------

__device__ inline float block_reduce_sum128(float val, float* red, int tid) {
    red[tid] = val; __syncthreads();
    for (int o = 64; o > 0; o >>= 1) { if (tid < o) red[tid] += red[tid + o]; __syncthreads(); }
    float r = red[0]; __syncthreads();
    return r;
}

__global__ __launch_bounds__(128)
void scan_kernel(const float* __restrict__ kn, const float* __restrict__ vv,
                 const float* __restrict__ qn, const float* __restrict__ alphas,
                 const float* __restrict__ promo, const float* __restrict__ fw,
                 const float* dw0, const float* dw1, const float* dw2,
                 const float* uw0, const float* uw1, const float* uw2,
                 float* __restrict__ y) {
    __shared__ float H0[64*64], H1[32*32], H2[16*16], H3[8*8];
    __shared__ float sdw0[32*64], sdw1[16*32], sdw2[8*16];
    __shared__ float suw0[64*32], suw1[64*16], suw2[64*8];
    __shared__ float kf[64], vf[64], qf[64];
    __shared__ float kns[64], vns[64], qns[64];
    __shared__ float al[120], pr[4], fwv[4];
    __shared__ float pred[64], kpj[64], outf[64], comb[64];
    __shared__ float red[128];

    const int b = blockIdx.x, tid = threadIdx.x;
    const int FD[4] = {64, 32, 16, 8};

    for (int i = tid; i < 64*64; i += 128) H0[i] = 0.0f;
    for (int i = tid; i < 32*32; i += 128) H1[i] = 0.0f;
    for (int i = tid; i < 16*16; i += 128) H2[i] = 0.0f;
    for (int i = tid; i <  8*8;  i += 128) H3[i] = 0.0f;
    for (int i = tid; i < 32*64; i += 128) sdw0[i] = dw0[i];
    for (int i = tid; i < 16*32; i += 128) sdw1[i] = dw1[i];
    for (int i = tid; i <  8*16; i += 128) sdw2[i] = dw2[i];
    for (int i = tid; i < 64*32; i += 128) suw0[i] = uw0[i];
    for (int i = tid; i < 64*16; i += 128) suw1[i] = uw1[i];
    for (int i = tid; i < 64*8;  i += 128) suw2[i] = uw2[i];
    __syncthreads();

    for (int s = 0; s < SEQ; ++s) {
        const size_t t = (size_t)b * SEQ + s;
        if (tid < 64) {
            kf[tid] = kn[t * 64 + tid];
            vf[tid] = vv[t * 64 + tid];
            qf[tid] = qn[t * 64 + tid];
            comb[tid] = 0.0f;
        }
        if (tid < 120) al[tid]  = alphas[t * 120 + tid];
        if (tid < 3)   pr[tid]  = promo[t * 3 + tid];
        if (tid < 4)   fwv[tid] = fw[t * 4 + tid];
        __syncthreads();

        int aoff = 0;
        for (int f = 0; f < 4; ++f) {
            const int d = FD[f];
            float* H = (f == 0) ? H0 : (f == 1) ? H1 : (f == 2) ? H2 : H3;

            // pred = q·H, kproj = k·H  (stride-d LDS columns: conflict-free)
            float errv = 0.0f;
            if (tid < d) {
                float ap = 0.0f, ak = 0.0f;
                for (int q = 0; q < d; ++q) {
                    float h = H[q * d + tid];
                    ap += qf[q] * h;
                    ak += kf[q] * h;
                }
                pred[tid] = ap; kpj[tid] = ak;
                float dvv = vf[tid] - ap;
                errv = dvv * dvv;
            }
            float err = block_reduce_sum128(errv, red, tid);
            float sur = sigmoidf_(err / (1.0f + 1e-6f));

            // H = a*H + (1-a) * sur * k (x) (v - kH)
            for (int idx = tid; idx < d * d; idx += 128) {
                int qi = idx / d, vi = idx - qi * d;
                float a = al[aoff + qi];
                H[idx] = a * H[idx] + (1.0f - a) * sur * kf[qi] * (vf[vi] - kpj[vi]);
            }
            __syncthreads();

            // out = q·H (post-update)
            if (tid < d) {
                float ao = 0.0f;
                for (int q = 0; q < d; ++q) ao += qf[q] * H[q * d + tid];
                outf[tid] = ao;
            }
            __syncthreads();

            // combine (fold 0 direct; folds 1-3 through up_w back to 64)
            if (f == 0) {
                if (tid < 64) comb[tid] += fwv[0] * outf[tid];
            } else {
                const float* uwf = (f == 1) ? suw0 : (f == 2) ? suw1 : suw2;
                if (tid < 64) {
                    float a2 = 0.0f;
                    for (int v2 = 0; v2 < d; ++v2) a2 += uwf[tid * d + v2] * outf[v2];
                    comb[tid] += fwv[f] * a2;
                }
            }

            // promotion to next fold
            if (f < 3) {
                const int d2 = FD[f + 1];
                const float* dwf = (f == 0) ? sdw0 : (f == 1) ? sdw1 : sdw2;
                float strength = pr[f] * sur;
                if (tid < d2) {
                    float kd = 0.0f, vd = 0.0f, qd = 0.0f;
                    for (int v2 = 0; v2 < d; ++v2) {
                        float w = dwf[tid * d + v2];
                        kd += kf[v2] * w; vd += vf[v2] * w; qd += qf[v2] * w;
                    }
                    kns[tid] = kd * strength;
                    vns[tid] = vd * strength;
                    qns[tid] = qd;
                }
                __syncthreads();
                float kk = (tid < d2) ? kns[tid] * kns[tid] : 0.0f;
                float knrm = fmaxf(sqrtf(block_reduce_sum128(kk, red, tid)), 1e-12f);
                float qq = (tid < d2) ? qns[tid] * qns[tid] : 0.0f;
                float qnrm = fmaxf(sqrtf(block_reduce_sum128(qq, red, tid)), 1e-12f);
                if (tid < d2) {
                    kf[tid] = kns[tid] / knrm;
                    vf[tid] = vns[tid];
                    qf[tid] = qns[tid] / qnrm;
                }
            }
            aoff += d;
            __syncthreads();
        }

        if (tid < 64) y[t * 64 + tid] = comb[tid];
        __syncthreads();
    }
}

// ---------- RMSNorm -> bf16 ----------

__global__ __launch_bounds__(64)
void rms_kernel(const float* __restrict__ y, const float* __restrict__ norm_w,
                bf16_t* __restrict__ yb) {
    __shared__ float red[64];
    int t = blockIdx.x, tid = threadIdx.x;
    float v = y[(size_t)t * 64 + tid];
    red[tid] = v * v; __syncthreads();
    for (int o = 32; o > 0; o >>= 1) { if (tid < o) red[tid] += red[tid + o]; __syncthreads(); }
    float scale = rsqrtf(red[0] * (1.0f / 64.0f) + 1e-6f);
    yb[(size_t)t * 64 + tid] = f2bf(v * scale * norm_w[tid]);
}

// ---------- launch ----------

extern "C" void kernel_launch(void* const* d_in, const int* in_sizes, int n_in,
                              void* d_out, int out_size, void* d_ws, size_t ws_size,
                              hipStream_t stream) {
    (void)in_sizes; (void)n_in; (void)out_size; (void)ws_size;

    const float* x        = (const float*)d_in[0];
    const float* Wk       = (const float*)d_in[1];
    const float* Wv       = (const float*)d_in[2];
    const float* Wq       = (const float*)d_in[3];
    const float* g0       = (const float*)d_in[4];
    const float* g1       = (const float*)d_in[5];
    const float* g2       = (const float*)d_in[6];
    const float* g3       = (const float*)d_in[7];
    const float* fgb0     = (const float*)d_in[8];
    const float* fgb1     = (const float*)d_in[9];
    const float* fgb2     = (const float*)d_in[10];
    const float* fgb3     = (const float*)d_in[11];
    const float* lam0     = (const float*)d_in[12];
    const float* lam1     = (const float*)d_in[13];
    const float* lam2     = (const float*)d_in[14];
    const float* lam3     = (const float*)d_in[15];
    const float* dw0      = (const float*)d_in[16];
    const float* dw1      = (const float*)d_in[17];
    const float* dw2      = (const float*)d_in[18];
    const float* uw0      = (const float*)d_in[19];
    const float* uw1      = (const float*)d_in[20];
    const float* uw2      = (const float*)d_in[21];
    const float* promo_w1 = (const float*)d_in[22];
    const float* promo_b1 = (const float*)d_in[23];
    const float* promo_w2 = (const float*)d_in[24];
    const float* promo_b2 = (const float*)d_in[25];
    const float* comb_w   = (const float*)d_in[26];
    const float* comb_b   = (const float*)d_in[27];
    const float* Wo       = (const float*)d_in[28];
    const float* norm_w   = (const float*)d_in[29];
    float* out = (float*)d_out;

    char* ws = (char*)d_ws;
    size_t off = 0;
    auto alloc = [&](size_t bytes) { void* p = ws + off; off += (bytes + 255) & ~(size_t)255; return p; };

    bf16_t* xb     = (bf16_t*)alloc((size_t)TOKENS * DMODEL * sizeof(bf16_t));
    bf16_t* Wall   = (bf16_t*)alloc((size_t)NCOL * DMODEL * sizeof(bf16_t));
    float*  P      = (float*) alloc((size_t)TOKENS * NCOL * sizeof(float));
    float*  kn     = (float*) alloc((size_t)TOKENS * 64 * sizeof(float));
    float*  vvb    = (float*) alloc((size_t)TOKENS * 64 * sizeof(float));
    float*  qn     = (float*) alloc((size_t)TOKENS * 64 * sizeof(float));
    float*  alphas = (float*) alloc((size_t)TOKENS * 120 * sizeof(float));
    float*  promo  = (float*) alloc((size_t)TOKENS * 3 * sizeof(float));
    float*  fw     = (float*) alloc((size_t)TOKENS * 4 * sizeof(float));
    float*  yraw   = (float*) alloc((size_t)TOKENS * 64 * sizeof(float));
    bf16_t* yb     = (bf16_t*)alloc((size_t)TOKENS * 64 * sizeof(bf16_t));
    bf16_t* Wob    = (bf16_t*)alloc((size_t)DMODEL * 64 * sizeof(bf16_t));

    // 1) x -> bf16
    {
        int n = TOKENS * DMODEL;
        convert_x_kernel<<<(n + 255) / 256, 256, 0, stream>>>(x, xb, n);
    }
    // 2) pack combined projection weights (bf16)
    {
        int n = NCOL * DMODEL;
        pack_wall_kernel<<<(n + 255) / 256, 256, 0, stream>>>(Wk, Wv, Wq, g0, g1, g2, g3,
                                                              promo_w1, comb_w, Wall);
    }
    // 3) pack Wo (bf16)
    {
        int n = DMODEL * 64;
        pack_wo_kernel<<<(n + 255) / 256, 256, 0, stream>>>(Wo, Wob, n);
    }
    // 4) GEMM1: P[8192,384] = xb[8192,2048] * Wall[384,2048]^T  (WMMA bf16)
    {
        int waves = (TOKENS / 16) * (NCOL / 16);     // 512 * 24 = 12288
        gemm_bf16_kernel<<<waves / 8, 256, 0, stream>>>(xb, Wall, P, TOKENS, NCOL, DMODEL);
    }
    // 5) elementwise post-process per token
    postproc_kernel<<<TOKENS, 128, 0, stream>>>(P, fgb0, fgb1, fgb2, fgb3,
                                                lam0, lam1, lam2, lam3,
                                                promo_b1, promo_w2, promo_b2, comb_b,
                                                kn, vvb, qn, alphas, promo, fw);
    // 6) sequential fold scan, one workgroup per batch, H-state LDS-resident
    scan_kernel<<<BATCH, 128, 0, stream>>>(kn, vvb, qn, alphas, promo, fw,
                                           dw0, dw1, dw2, uw0, uw1, uw2, yraw);
    // 7) RMSNorm -> bf16
    rms_kernel<<<TOKENS, 64, 0, stream>>>(yraw, norm_w, yb);
    // 8) GEMM2: out[8192,2048] = yb[8192,64] * Wob[2048,64]^T  (WMMA bf16, K=64)
    {
        int waves = (TOKENS / 16) * (DMODEL / 16);   // 512 * 128 = 65536
        gemm_bf16_kernel<<<waves / 8, 256, 0, stream>>>(yb, Wob, out, TOKENS, DMODEL, 64);
    }
}
